// VNInvAttentionLayer_71708773974446
// MI455X (gfx1250) — matrix operand order repaired and checked by reading, not compiled, not run on hardware
//
#include <hip/hip_runtime.h>
#include <hip/hip_bf16.h>
#include <stdint.h>

// Problem constants (fixed by the reference)
#define B_   4
#define D_   512
#define L_   1024
#define H_   8
#define DH_  64
#define S_   1024

typedef __bf16   v16bf __attribute__((ext_vector_type(16)));
typedef float    v8f   __attribute__((ext_vector_type(8)));
typedef uint32_t v4u   __attribute__((ext_vector_type(4)));
typedef int      i32x4 __attribute__((ext_vector_type(4)));
typedef int      i32x8 __attribute__((ext_vector_type(8)));

union AB16 { v16bf v; v4u q[2]; uint32_t u[8]; };

// ---- CDNA5 data-movement tier selection ----
// Tier 1: TDM builtin; Tier 2: async builtin; Tier 3: inline-asm async (always).
#if defined(__has_builtin)
#if __has_builtin(__builtin_amdgcn_tensor_load_to_lds) && __has_builtin(__builtin_amdgcn_s_wait_tensorcnt)
#define USE_TDM 1
#endif
#if __has_builtin(__builtin_amdgcn_global_load_async_to_lds_b128) && __has_builtin(__builtin_amdgcn_s_wait_asynccnt)
#define USE_ASYNC 1
#endif
#endif
#ifndef USE_TDM
#define USE_TDM 0
#endif
#ifndef USE_ASYNC
#define USE_ASYNC 0
#endif

#if USE_TDM
#warning "probe: USE_TDM=1 (tensor_load_to_lds builtin active)"
#elif USE_ASYNC
#warning "probe: USE_ASYNC=1 (global_load_async_to_lds_b128 builtin active)"
#else
#warning "probe: inline-asm async tier active (no TDM/async builtins found)"
#endif

#if USE_ASYNC
#define AS1 __attribute__((address_space(1)))
#define AS3 __attribute__((address_space(3)))
#endif

__device__ __forceinline__ uint16_t f2bf(float f) {
  uint32_t u = __float_as_uint(f);
  u += 0x7FFFu + ((u >> 16) & 1u);   // round-to-nearest-even bf16
  return (uint16_t)(u >> 16);
}

// Two aligned 16B LDS loads -> one 16x32 bf16 WMMA operand.
// A-operand pattern: chunks at p and p+8 elements.  B (K-contiguous): p and p+8.
__device__ __forceinline__ v16bf lds_op_split(const uint16_t* p) {
  AB16 a; a.q[0] = *(const v4u*)p; a.q[1] = *(const v4u*)(p + 8); return a.v;
}
__device__ __forceinline__ v16bf lds_op_contig(const uint16_t* p) {
  AB16 a; a.q[0] = *(const v4u*)p; a.q[1] = *(const v4u*)(p + 8); return a.v;
}

#if USE_TDM
// TDM 2D tile load: `rows` x 64 bf16, row stride 64 elements, into LDS.
// D# bitfields per CDNA5 ISA.
__device__ __forceinline__ void tdm_load_k64(uint32_t ldsOff, const void* gaddr,
                                             uint32_t rows) {
  uint64_t ga = (uint64_t)(uintptr_t)gaddr;
  v4u g0;
  g0[0] = 1u;                                                // count=1, user D#
  g0[1] = ldsOff;                                            // lds_addr (bytes)
  g0[2] = (uint32_t)ga;                                      // global_addr[31:0]
  g0[3] = (uint32_t)((ga >> 32) & 0x01FFFFFFu) | (2u << 30); // addr[56:32], type=2
  i32x8 g1;
  g1[0] = (int)(1u << 16);                 // workgroup_mask=0, data_size=1 (2B)
  g1[1] = (int)((uint32_t)DH_ << 16);      // tensor_dim0 = 64 (bits 63:48)
  g1[2] = (int)((uint32_t)L_ << 16);       // dim0 hi=0 | tensor_dim1 lo16 (95:80)
  g1[3] = (int)((uint32_t)DH_ << 16);      // dim1 hi=0 | tile_dim0 = 64 (127:112)
  g1[4] = (int)rows;                       // tile_dim1; tile_dim2=0
  g1[5] = (int)DH_;                        // tensor_dim0_stride lo32 = 64
  g1[6] = 0;
  g1[7] = 0;
  i32x4 z4 = {0, 0, 0, 0};                 // groups 2/3 unused (2D tensor)
#if __clang_major__ >= 23
  i32x8 z8 = {0, 0, 0, 0, 0, 0, 0, 0};
  __builtin_amdgcn_tensor_load_to_lds(g0, g1, z4, z4, z8, 0);
#else
  __builtin_amdgcn_tensor_load_to_lds(g0, g1, z4, z4, 0);
#endif
}
#endif

#if !USE_TDM && !USE_ASYNC
// Inline-asm async copy (ISA §10.7 GLOBAL_LOAD_ASYNC_TO_LDS_B128, ASYNCcnt-tracked).
// First operand = per-lane LDS byte address (VDST), second = 64-bit global address.
__device__ __forceinline__ void asm_async_b128(uint32_t ldsAddr, const void* g) {
  uint64_t ga = (uint64_t)(uintptr_t)g;
  asm volatile("global_load_async_to_lds_b128 %0, %1, off"
               :: "v"(ldsAddr), "v"(ga)
               : "memory");
}
#endif

// Issue the staged copy of one 32x64 bf16 K chunk (4 KB) into LDS.
__device__ __forceinline__ void k_chunk_issue(const uint16_t* src, uint16_t* dstLds,
                                              int tid) {
#if USE_TDM
  if ((tid >> 5) == 0)
    tdm_load_k64((uint32_t)(uintptr_t)(void*)dstLds, src, 32);
#elif USE_ASYNC
  // 256 threads x 16 B = 4 KB; one async b128 instruction per wave.
  __builtin_amdgcn_global_load_async_to_lds_b128(
      (const AS1 char*)(const void*)src + tid * 16,
      (AS3 char*)(void*)dstLds + tid * 16, 0, 0);
#else
  asm_async_b128((uint32_t)(uintptr_t)(void*)dstLds + (uint32_t)tid * 16,
                 (const char*)src + tid * 16);
#endif
}
// Wait for the previous chunk (one more may remain in flight when more==true).
__device__ __forceinline__ void k_chunk_wait(bool more, int tid) {
#if USE_TDM
  if ((tid >> 5) == 0) {
    if (more) __builtin_amdgcn_s_wait_tensorcnt((short)1);
    else      __builtin_amdgcn_s_wait_tensorcnt((short)0);
  }
#elif USE_ASYNC
  if (more) __builtin_amdgcn_s_wait_asynccnt((short)1);
  else      __builtin_amdgcn_s_wait_asynccnt((short)0);
#else
  (void)tid;
  if (more) asm volatile("s_wait_asynccnt 0x1" ::: "memory");
  else      asm volatile("s_wait_asynccnt 0x0" ::: "memory");
#endif
}

// ---------------------------------------------------------------------------
// fp32 -> bf16 weight conversion
// ---------------------------------------------------------------------------
__global__ void cvt_bf16_kernel(const float* __restrict__ a,
                                uint16_t* __restrict__ o, int nEl) {
  int i = blockIdx.x * blockDim.x + threadIdx.x;
  if (i < nEl) o[i] = f2bf(a[i]);
}

// ---------------------------------------------------------------------------
// Projection:  y[b,l,e,v] = sum_d W[e,d] * x[b,d,v,l]
// Block tile 64(e) x 32(l); K-loop over d in steps of 32.
// Staging fully unrolled: all global loads issue before any LDS store.
// ---------------------------------------------------------------------------
__global__ __launch_bounds__(256) void proj_kernel(
    const float* __restrict__ x,        // [B, D, 3, L]
    const uint16_t* __restrict__ wbf,   // [D, D] bf16
    uint16_t* __restrict__ yp) {        // [B, H, 3, L, DH] bf16
  __shared__ alignas(16) uint16_t WL[64 * 32];  // [e][d]
  __shared__ alignas(16) uint16_t XL[32 * 32];  // [l][d] (transposed at staging)

  const int tid  = threadIdx.x;
  const int lane = tid & 31, wid = tid >> 5;
  const int wm = wid >> 1, wn = wid & 1;
  const int n = lane & 15, half = lane >> 4;

  const int eBase = (blockIdx.x >> 5) * 64;
  const int lBase = (blockIdx.x & 31) * 32;
  const int b = blockIdx.y / 3, vv = blockIdx.y % 3;

  v8f acc = {};

  for (int kk = 0; kk < D_; kk += 32) {
    uint32_t wreg[4];
    const uint32_t* wsrc = (const uint32_t*)wbf;
#pragma unroll
    for (int u = 0; u < 4; ++u) {
      int t = tid + 256 * u;
      wreg[u] = wsrc[(size_t)(eBase + (t >> 4)) * (D_ / 2) + (kk >> 1) + (t & 15)];
    }
    float xreg[4];
#pragma unroll
    for (int u = 0; u < 4; ++u) {
      int t = tid + 256 * u;
      xreg[u] = x[((size_t)(b * D_ + kk + (t >> 5)) * 3 + vv) * L_ + lBase + (t & 31)];
    }
    uint32_t* wl = (uint32_t*)WL;
#pragma unroll
    for (int u = 0; u < 4; ++u) wl[tid + 256 * u] = wreg[u];
#pragma unroll
    for (int u = 0; u < 4; ++u) {
      int t = tid + 256 * u;
      XL[(t & 31) * 32 + (t >> 5)] = f2bf(xreg[u]);  // transpose: [l][d]
    }
    __syncthreads();

    v16bf a = lds_op_split(WL + (wm * 16 + n) * 32 + half * 16);
    v16bf bbv = lds_op_contig(XL + (wn * 16 + n) * 32 + half * 16);
    acc = __builtin_amdgcn_wmma_f32_16x16x32_bf16(false, a, false, bbv,
                                                  (short)0, acc, false, false);
    __syncthreads();
  }

  // C layout: VGPR r -> row M = r + 8*half, col N = n; dh consecutive over r
  const int l = lBase + wn * 16 + n;
#pragma unroll
  for (int r = 0; r < 8; ++r) {
    int e = eBase + wm * 16 + r + 8 * half;
    int h = e >> 6, dh = e & 63;
    yp[(((size_t)(b * H_ + h) * 3 + vv) * L_ + l) * DH_ + dh] = f2bf(acc[r]);
  }
}

// ---------------------------------------------------------------------------
// Fused attention per (b,h,v).  Block = 8 waves, 128 Q rows.
// Pass 1: streaming online max/sum over S (K chunks async/TDM double-buffered).
// Pass 2: recompute scores, write normalized attn exactly once, accumulate
//         O = attn @ V (attn tile re-laid out via per-wave LDS patch; V staged
//         transposed so the AV B-operand is K-contiguous).
// ---------------------------------------------------------------------------
__global__ __launch_bounds__(256) void attn_kernel(
    const uint16_t* __restrict__ qp,   // [B,H,3,L,DH] bf16
    const uint16_t* __restrict__ kp,
    const uint16_t* __restrict__ vp,
    float* __restrict__ attn,          // [B,H,3,L,S] f32 (output region)
    uint16_t* __restrict__ op) {       // [B,L,3,D] bf16
  __shared__ alignas(16) uint16_t QL[128 * 64];    // Q block [l][dh]
  __shared__ alignas(16) uint16_t KL[2][32 * 64];  // K chunk [s][dh], double-buffered
  __shared__ alignas(16) uint16_t VLt[64 * 32];    // V chunk TRANSPOSED [dh][s]
  __shared__ alignas(16) uint16_t PL[8][16 * 32];  // per-wave attn patch [l][s]

  const int tid  = threadIdx.x;
  const int lane = tid & 31, wid = tid >> 5;
  const int n = lane & 15, half = lane >> 4;

  const int y = blockIdx.y;
  const int b = y / (H_ * 3);
  const int rem = y % (H_ * 3);
  const int h = rem / 3, vv = rem % 3;
  const int blockL = blockIdx.x * 128;

  const size_t base = ((size_t)((b * H_ + h) * 3 + vv)) * L_ * DH_;
  const float kScale = 0.57735026918962576f;  // 1/sqrt(3)

  // stage Q block (16 KB, fully contiguous)
  {
    const v4u* qsrc = (const v4u*)(qp + base + (size_t)blockL * DH_);
    v4u* ql = (v4u*)QL;
    v4u qreg[4];
#pragma unroll
    for (int u = 0; u < 4; ++u) qreg[u] = qsrc[tid + 256 * u];
#pragma unroll
    for (int u = 0; u < 4; ++u) ql[tid + 256 * u] = qreg[u];
  }
  __syncthreads();

  // Wave's Q A-operands: a0 = dh 0..31, a1 = dh 32..63 (live across both passes)
  const uint16_t* qpx = QL + (wid * 16 + n) * 64 + half * 16;
  v16bf a0 = lds_op_split(qpx);
  v16bf a1 = lds_op_split(qpx + 32);

  float rm[8], rs[8];
#pragma unroll
  for (int r = 0; r < 8; ++r) { rm[r] = -3.0e38f; rs[r] = 0.0f; }

  const int SCH = S_ / 32;

  // ---------------- pass 1: streaming row max / sum ----------------
  k_chunk_issue(kp + base, KL[0], tid);
  for (int c = 0; c < SCH; ++c) {
    uint16_t* KC = KL[c & 1];
    if (c + 1 < SCH)
      k_chunk_issue(kp + base + (size_t)((c + 1) * 32) * DH_, KL[(c + 1) & 1], tid);
    k_chunk_wait(c + 1 < SCH, tid);
    __syncthreads();

    v8f s0 = {}, s1 = {};
#pragma unroll
    for (int sub = 0; sub < 2; ++sub) {
      // B = K^T (K-dim = dh, N = s): lane n -> s row; dh contiguous in LDS
      const uint16_t* bp = KC + (sub * 16 + n) * 64 + half * 16;
      v16bf b0 = lds_op_contig(bp);
      v16bf b1 = lds_op_contig(bp + 32);
      v8f acc = {};
      acc = __builtin_amdgcn_wmma_f32_16x16x32_bf16(false, a0, false, b0,
                                                    (short)0, acc, false, false);
      acc = __builtin_amdgcn_wmma_f32_16x16x32_bf16(false, a1, false, b1,
                                                    (short)0, acc, false, false);
      if (sub == 0) s0 = acc; else s1 = acc;
    }

#pragma unroll
    for (int r = 0; r < 8; ++r) {
      float v0 = s0[r] * kScale, v1 = s1[r] * kScale;
      float cm = fmaxf(v0, v1);
#pragma unroll
      for (int m = 8; m >= 1; m >>= 1)
        cm = fmaxf(cm, __shfl_xor(cm, m, 32));   // reduce over 16 cols (same half)
      float nm = fmaxf(rm[r], cm);
      float cs = __expf(v0 - nm) + __expf(v1 - nm);
#pragma unroll
      for (int m = 8; m >= 1; m >>= 1)
        cs += __shfl_xor(cs, m, 32);
      rs[r] = rs[r] * __expf(rm[r] - nm) + cs;
      rm[r] = nm;
    }
    __syncthreads();
  }

  float inv[8];
#pragma unroll
  for (int r = 0; r < 8; ++r) inv[r] = 1.0f / rs[r];

  // ---------------- pass 2: normalize, emit attn, accumulate O ----------------
  v8f o[4] = {};
  const size_t attnBase = ((size_t)((b * H_ + h) * 3 + vv)) * L_ * S_;

  k_chunk_issue(kp + base, KL[0], tid);
  for (int c = 0; c < SCH; ++c) {
    uint16_t* KC = KL[c & 1];
    // stage V transposed: VLt[dh][s] <- vp[.., c*32+s, dh] (load phase first)
    uint32_t vreg[4];
#pragma unroll
    for (int u = 0; u < 4; ++u) {
      int t = tid + 256 * u;
      vreg[u] = *(const uint32_t*)(vp + base + (size_t)(c * 32 + (t >> 5)) * DH_ +
                                   2 * (t & 31));
    }
#pragma unroll
    for (int u = 0; u < 4; ++u) {
      int t = tid + 256 * u;
      int s = t >> 5, dp = t & 31;
      VLt[(2 * dp) * 32 + s]     = (uint16_t)vreg[u];
      VLt[(2 * dp + 1) * 32 + s] = (uint16_t)(vreg[u] >> 16);
    }
    if (c + 1 < SCH)
      k_chunk_issue(kp + base + (size_t)((c + 1) * 32) * DH_, KL[(c + 1) & 1], tid);
    k_chunk_wait(c + 1 < SCH, tid);
    __syncthreads();

#pragma unroll
    for (int sub = 0; sub < 2; ++sub) {
      const uint16_t* bp = KC + (sub * 16 + n) * 64 + half * 16;
      v16bf b0 = lds_op_contig(bp);
      v16bf b1 = lds_op_contig(bp + 32);
      v8f acc = {};
      acc = __builtin_amdgcn_wmma_f32_16x16x32_bf16(false, a0, false, b0,
                                                    (short)0, acc, false, false);
      acc = __builtin_amdgcn_wmma_f32_16x16x32_bf16(false, a1, false, b1,
                                                    (short)0, acc, false, false);
#pragma unroll
      for (int r = 0; r < 8; ++r) {
        float p = __expf(acc[r] * kScale - rm[r]) * inv[r];
        int row = r + 8 * half;
        int l = blockL + wid * 16 + row;
        int s = c * 32 + sub * 16 + n;
        attn[attnBase + (size_t)l * S_ + s] = p;       // exact attn output, once
        PL[wid][row * 32 + sub * 16 + n] = f2bf(p);    // re-layout via LDS
      }
    }
    __syncthreads();

    // O += attn_tile(16x32) @ V_chunk(32x64): 4 N-subtiles of 16
    v16bf pa = lds_op_split(&PL[wid][n * 32 + half * 16]);
#pragma unroll
    for (int nd = 0; nd < 4; ++nd) {
      v16bf vb = lds_op_contig(VLt + (nd * 16 + n) * 32 + half * 16);
      o[nd] = __builtin_amdgcn_wmma_f32_16x16x32_bf16(false, pa, false, vb,
                                                      (short)0, o[nd], false, false);
    }
    __syncthreads();
  }

  // Store O (bf16) into [B, L, 3, D] workspace for the output projection
#pragma unroll
  for (int nd = 0; nd < 4; ++nd) {
#pragma unroll
    for (int r = 0; r < 8; ++r) {
      int l = blockL + wid * 16 + r + 8 * half;
      int dcol = h * 64 + nd * 16 + n;
      op[((size_t)(b * L_ + l) * 3 + vv) * D_ + dcol] = f2bf(o[nd][r]);
    }
  }
}

// ---------------------------------------------------------------------------
// Output projection: out[b,e,v,l] = sum_d Wo[e,d] * op[b,l,v,d]
// ---------------------------------------------------------------------------
__global__ __launch_bounds__(256) void outproj_kernel(
    const uint16_t* __restrict__ wbf,  // Wo bf16 [D,D]
    const uint16_t* __restrict__ op,   // [B,L,3,D] bf16
    float* __restrict__ out) {         // [B,D,3,L] f32
  __shared__ alignas(16) uint16_t WL[64 * 32];  // [e][d]
  __shared__ alignas(16) uint16_t OL[32 * 32];  // [l][d]

  const int tid  = threadIdx.x;
  const int lane = tid & 31, wid = tid >> 5;
  const int wm = wid >> 1, wn = wid & 1;
  const int n = lane & 15, half = lane >> 4;

  const int eBase = (blockIdx.x >> 5) * 64;
  const int lBase = (blockIdx.x & 31) * 32;
  const int b = blockIdx.y / 3, vv = blockIdx.y % 3;

  v8f acc = {};

  for (int kk = 0; kk < D_; kk += 32) {
    uint32_t wreg[4];
    const uint32_t* wsrc = (const uint32_t*)wbf;
#pragma unroll
    for (int u = 0; u < 4; ++u) {
      int t = tid + 256 * u;
      wreg[u] = wsrc[(size_t)(eBase + (t >> 4)) * (D_ / 2) + (kk >> 1) + (t & 15)];
    }
    uint32_t oreg[2];
#pragma unroll
    for (int u = 0; u < 2; ++u) {
      int t = tid + 256 * u;
      oreg[u] = *(const uint32_t*)(op + ((size_t)(b * L_ + lBase + (t >> 4)) * 3 + vv) * D_ +
                                   kk + 2 * (t & 15));
    }
    uint32_t* wl = (uint32_t*)WL;
#pragma unroll
    for (int u = 0; u < 4; ++u) wl[tid + 256 * u] = wreg[u];
    uint32_t* ol = (uint32_t*)OL;
#pragma unroll
    for (int u = 0; u < 2; ++u) ol[tid + 256 * u] = oreg[u];
    __syncthreads();

    v16bf a = lds_op_split(WL + (wm * 16 + n) * 32 + half * 16);
    v16bf bbv = lds_op_contig(OL + (wn * 16 + n) * 32 + half * 16);
    acc = __builtin_amdgcn_wmma_f32_16x16x32_bf16(false, a, false, bbv,
                                                  (short)0, acc, false, false);
    __syncthreads();
  }

  const int l = lBase + wn * 16 + n;
#pragma unroll
  for (int r = 0; r < 8; ++r) {
    int e = eBase + wm * 16 + r + 8 * half;
    out[((size_t)(b * D_ + e) * 3 + vv) * L_ + l] = acc[r];
  }
}

// ---------------------------------------------------------------------------
extern "C" void kernel_launch(void* const* d_in, const int* in_sizes, int n_in,
                              void* d_out, int out_size, void* d_ws, size_t ws_size,
                              hipStream_t stream) {
  (void)in_sizes; (void)n_in; (void)out_size; (void)ws_size;

  const float* q  = (const float*)d_in[0];
  const float* k  = (const float*)d_in[1];
  const float* v  = (const float*)d_in[2];
  const float* Wq = (const float*)d_in[3];
  const float* Wk = (const float*)d_in[4];
  const float* Wv = (const float*)d_in[5];
  const float* Wo = (const float*)d_in[6];
  // d_in[7] = n_heads (== 8, baked into constants)

  // Workspace (bf16), ~50 MB: 4 weights (0.5 MB each) + qp/kp/vp/op (12 MB each)
  uint16_t* wqb = (uint16_t*)d_ws;
  uint16_t* wkb = wqb + (size_t)D_ * D_;
  uint16_t* wvb = wkb + (size_t)D_ * D_;
  uint16_t* wob = wvb + (size_t)D_ * D_;
  uint16_t* qp  = wob + (size_t)D_ * D_;
  const size_t pElems = (size_t)B_ * H_ * 3 * L_ * DH_;
  uint16_t* kp = qp + pElems;
  uint16_t* vp = kp + pElems;
  uint16_t* op = vp + pElems;

  float* outp  = (float*)d_out;                       // [B,D,3,L]
  float* attnp = outp + (size_t)B_ * D_ * 3 * L_;     // [B,H,3,L,S]

  dim3 blk(256);

  {
    int nEl = D_ * D_;
    dim3 g((nEl + 255) / 256);
    cvt_bf16_kernel<<<g, blk, 0, stream>>>(Wq, wqb, nEl);
    cvt_bf16_kernel<<<g, blk, 0, stream>>>(Wk, wkb, nEl);
    cvt_bf16_kernel<<<g, blk, 0, stream>>>(Wv, wvb, nEl);
    cvt_bf16_kernel<<<g, blk, 0, stream>>>(Wo, wob, nEl);
  }

  dim3 gp(256, B_ * 3);  // 8 e-tiles * 32 l-tiles; (b, v) on y
  proj_kernel<<<gp, blk, 0, stream>>>(q, wqb, qp);
  proj_kernel<<<gp, blk, 0, stream>>>(k, wkb, kp);
  proj_kernel<<<gp, blk, 0, stream>>>(v, wvb, vp);

  dim3 ga(L_ / 128, B_ * H_ * 3);  // 8 x 96
  attn_kernel<<<ga, blk, 0, stream>>>(qp, kp, vp, attnp, op);

  outproj_kernel<<<gp, blk, 0, stream>>>(wob, op, outp);
}
